// Decoder_59760174957314
// MI455X (gfx1250) — matrix operand order repaired and checked by reading, not compiled, not run on hardware
//
#include <hip/hip_runtime.h>
#include <hip/hip_bf16.h>

// Problem constants (match reference): B=4, N=1024, E=16, H=32
#define NB    4
#define NN    1024
#define NE    16
#define NH    32
#define NROWS (NB * NN)        // 4096 node rows

typedef __attribute__((ext_vector_type(16))) _Float16 v16h;
typedef __attribute__((ext_vector_type(8)))  float    v8f;
typedef __attribute__((ext_vector_type(4)))  float    f4v;

// ---------------------------------------------------------------------------
// Stage 1: node projections via WMMA.
//   HI[r, :] = emb[r, :] @ W1[0:16, :]      (A has emb in K in [0,16))
//   HJ[r, :] = emb[r, :] @ W1[16:32, :]     (A has emb in K in [16,32))
// One wave per 16-row M-tile. K = 2E = 32 -> exactly one v_wmma_f32_16x16x32_f16
// step per (A-variant, N-tile). f16 inputs (unit-scale data), fp32 accumulate.
// ---------------------------------------------------------------------------
__global__ __launch_bounds__(32) void node_proj_wmma(
    const float* __restrict__ emb,   // [NROWS, NE]
    const float* __restrict__ W1,    // [2*NE, NH] row-major
    float* __restrict__ HI,          // [NROWS, NH]
    float* __restrict__ HJ)          // [NROWS, NH]
{
    const int lane = threadIdx.x;        // 0..31, full wave (EXEC all 1s)
    const int g    = lane >> 4;          // lane group 0/1
    const int m    = lane & 15;          // row within tile / column index
    const int row  = blockIdx.x * 16 + m;

    // This lane's 8 emb values: columns [8g, 8g+8)
    const float* ep = emb + row * NE + g * 8;
    f4v e_lo = *(const f4v*)ep;
    f4v e_hi = *(const f4v*)(ep + 4);
    float e8[8] = { e_lo.x, e_lo.y, e_lo.z, e_lo.w, e_hi.x, e_hi.y, e_hi.z, e_hi.w };

    // A-matrix 16x32 f16 layout (ISA 7.12.2): lane holds row m = lane%16;
    //   element e<8  -> K = 8g + e          (covers K in [0,16))
    //   element e>=8 -> K = 16 + 8g + (e-8) (covers K in [16,32))
    v16h a_lo = {};   // emb in K<16 half  -> multiplies W1 rows 0..15  => hi
    v16h a_hi = {};   // emb in K>=16 half -> multiplies W1 rows 16..31 => hj
#pragma unroll
    for (int e = 0; e < 8; ++e) {
        _Float16 h = (_Float16)e8[e];
        a_lo[e]     = h;   // K = 8g+e        -> emb col 8g+e
        a_hi[e + 8] = h;   // K = 16 + 8g + e -> emb col 8g+e
    }

    // B-matrix 32x16 f16 layout (same K order as A): element e<8 -> K = 8g + e,
    // element e>=8 -> K = 16 + 8g + (e-8), column n = lane%16.
    v16h b0, b1v;   // N-tiles: W1 columns [0,16) and [16,32)
#pragma unroll
    for (int e = 0; e < 16; ++e) {
        const int k = (e < 8) ? (8 * g + e) : (16 + 8 * g + (e - 8));
        b0[e]  = (_Float16)W1[k * NH + m];
        b1v[e] = (_Float16)W1[k * NH + 16 + m];
    }

    const v8f cz = {};
    v8f d_hi0 = __builtin_amdgcn_wmma_f32_16x16x32_f16(false, a_lo, false, b0,  (short)0, cz, false, false);
    v8f d_hi1 = __builtin_amdgcn_wmma_f32_16x16x32_f16(false, a_lo, false, b1v, (short)0, cz, false, false);
    v8f d_hj0 = __builtin_amdgcn_wmma_f32_16x16x32_f16(false, a_hi, false, b0,  (short)0, cz, false, false);
    v8f d_hj1 = __builtin_amdgcn_wmma_f32_16x16x32_f16(false, a_hi, false, b1v, (short)0, cz, false, false);

    // D layout: c[r] = D[8g + r, n = lane%16]
    const int orow0 = blockIdx.x * 16 + 8 * g;
#pragma unroll
    for (int r = 0; r < 8; ++r) {
        const int orow = orow0 + r;
        HI[orow * NH + m]      = d_hi0[r];
        HI[orow * NH + 16 + m] = d_hi1[r];
        HJ[orow * NH + m]      = d_hj0[r];
        HJ[orow * NH + 16 + m] = d_hj1[r];
    }
}

// ---------------------------------------------------------------------------
// Stage 2: out[b,i,j] = sigmoid(b2 + sum_h relu(HI[b,i,h]+b1[h] + HJ[b,j,h]) * W2[h])
// Tiling: 32 i x 64 j per 256-thread block. hi rows (b1 pre-folded) live in
// 32 VGPRs/thread; hj tile in LDS with 36-dword row stride (16B aligned b128
// loads, conflict-free bank quads). Each thread -> 1 i x 8 contiguous j.
// ---------------------------------------------------------------------------
#define TI 32
#define TJ 64
#define LSTR 36

__device__ __forceinline__ float sigmoidf_fast(float x) {
    return __builtin_amdgcn_rcpf(1.0f + __expf(-x));
}

__global__ __launch_bounds__(256) void pair_mlp(
    const float* __restrict__ HI, const float* __restrict__ HJ,
    const float* __restrict__ b1, const float* __restrict__ W2,
    const float* __restrict__ b2, float* __restrict__ out)
{
    __shared__ float sHI[TI * LSTR];
    __shared__ float sHJ[TJ * LSTR];
    __shared__ float sW2[NH];

    const int t  = threadIdx.x;
    const int bj = blockIdx.x * TJ;
    const int bi = blockIdx.y * TI;
    const size_t nodeBase = (size_t)blockIdx.z * NN;

    if (t < NH) sW2[t] = W2[t];

    // sHI: TI*32 = 1024 floats, 4/thread, fold in b1
    {
        const int elem = t * 4;
        const int r = elem >> 5, c = elem & 31;
        f4v v  = *(const f4v*)&HI[(nodeBase + bi + r) * NH + c];
        f4v bv = *(const f4v*)&b1[c];
        v.x += bv.x; v.y += bv.y; v.z += bv.z; v.w += bv.w;
        *(f4v*)&sHI[r * LSTR + c] = v;
    }
    // sHJ: TJ*32 = 2048 floats, 8/thread
#pragma unroll
    for (int p = 0; p < 2; ++p) {
        const int elem = (t + p * 256) * 4;
        const int r = elem >> 5, c = elem & 31;
        *(f4v*)&sHJ[r * LSTR + c] = *(const f4v*)&HJ[(nodeBase + bj + r) * NH + c];
    }
    __syncthreads();

    const int ti = t >> 3;         // i within tile: 0..31
    const int tj = (t & 7) * 8;    // j offset within tile: 0,8,...,56

    float hi_r[NH];
#pragma unroll
    for (int k = 0; k < NH / 4; ++k)
        *(f4v*)&hi_r[k * 4] = *(const f4v*)&sHI[ti * LSTR + k * 4];

    float acc[8] = {0.f, 0.f, 0.f, 0.f, 0.f, 0.f, 0.f, 0.f};
#pragma unroll
    for (int h4 = 0; h4 < NH / 4; ++h4) {
        const f4v w = *(const f4v*)&sW2[h4 * 4];
        const float a0 = hi_r[h4 * 4 + 0];
        const float a1 = hi_r[h4 * 4 + 1];
        const float a2 = hi_r[h4 * 4 + 2];
        const float a3 = hi_r[h4 * 4 + 3];
#pragma unroll
        for (int jj = 0; jj < 8; ++jj) {
            const f4v hj = *(const f4v*)&sHJ[(tj + jj) * LSTR + h4 * 4];
            acc[jj] += fmaxf(a0 + hj.x, 0.f) * w.x
                     + fmaxf(a1 + hj.y, 0.f) * w.y
                     + fmaxf(a2 + hj.z, 0.f) * w.z
                     + fmaxf(a3 + hj.w, 0.f) * w.w;
        }
    }

    const float bias2 = b2[0];
    float* op = out + (nodeBase + bi + ti) * (size_t)NN + (bj + tj);
#pragma unroll
    for (int q = 0; q < 2; ++q) {
        f4v o;
        o.x = sigmoidf_fast(acc[q * 4 + 0] + bias2);
        o.y = sigmoidf_fast(acc[q * 4 + 1] + bias2);
        o.z = sigmoidf_fast(acc[q * 4 + 2] + bias2);
        o.w = sigmoidf_fast(acc[q * 4 + 3] + bias2);
        *(f4v*)(op + q * 4) = o;
    }
}

// ---------------------------------------------------------------------------
extern "C" void kernel_launch(void* const* d_in, const int* in_sizes, int n_in,
                              void* d_out, int out_size, void* d_ws, size_t ws_size,
                              hipStream_t stream) {
    (void)in_sizes; (void)n_in; (void)out_size; (void)ws_size;

    const float* emb = (const float*)d_in[0];  // [4,1024,16]
    const float* W1  = (const float*)d_in[1];  // [32,32]
    const float* b1  = (const float*)d_in[2];  // [32]
    const float* W2  = (const float*)d_in[3];  // [32]
    const float* b2  = (const float*)d_in[4];  // [1]
    float* out = (float*)d_out;                // [4,1024,1024]

    float* HI = (float*)d_ws;                  // [4096,32]
    float* HJ = HI + (size_t)NROWS * NH;       // [4096,32]

    node_proj_wmma<<<dim3(NROWS / 16), dim3(32), 0, stream>>>(emb, W1, HI, HJ);
    pair_mlp<<<dim3(NN / TJ, NN / TI, NB), dim3(256), 0, stream>>>(HI, HJ, b1, W2, b2, out);
}
